// AE_skipLSTM_18915035971749
// MI455X (gfx1250) — matrix operand order repaired and checked by reading, not compiled, not run on hardware
//
#include <hip/hip_runtime.h>
#include <math.h>

// ---------------------------------------------------------------------------
// CDNA5 wave32 WMMA f16 helpers (v_wmma_f32_16x16x32_f16)
// ---------------------------------------------------------------------------
typedef __attribute__((ext_vector_type(16))) _Float16 v16h;
typedef __attribute__((ext_vector_type(8)))  float    v8f;

__device__ __forceinline__ v8f wmma_f16(v16h a, v16h b, v8f c) {
  // 8 args: (neg_a, A, neg_b, B, c_mod, C, reuse_a, reuse_b)
  return __builtin_amdgcn_wmma_f32_16x16x32_f16(false, a, false, b, (short)0, c,
                                                false, false);
}

// A fragment (16x32 f16). lane holds row m = (lane&15).
// lanes 0-15:  e=0..15 -> K=0..15
// lanes 16-31: e=0..15 -> K=16..31
__device__ __forceinline__ v16h a_frag_f32(const float* __restrict__ p, int lane) {
  const int kh = (lane >> 4) << 4;
  v16h r;
#pragma unroll
  for (int i = 0; i < 8; ++i) {
    r[i]     = (_Float16)p[kh + i];
    r[i + 8] = (_Float16)p[8 + kh + i];
  }
  return r;
}
__device__ __forceinline__ v16h a_frag_f16(const _Float16* __restrict__ p, int lane) {
  const int kh = (lane >> 4) << 4;
  v16h r;
#pragma unroll
  for (int i = 0; i < 8; ++i) {
    r[i]     = p[kh + i];
    r[i + 8] = p[8 + kh + i];
  }
  return r;
}

// B fragment (32x16). lane holds col n = (lane&15); k = ((lane>>4)<<4) + e.
// caller passes p = B + (kbase + ((lane>>4)<<4))*ldb + n
__device__ __forceinline__ v16h b_frag_f32_strided(const float* __restrict__ p, int ldb) {
  v16h r;
#pragma unroll
  for (int e = 0; e < 16; ++e) r[e] = (_Float16)p[(size_t)e * ldb];
  return r;
}
__device__ __forceinline__ v16h b_frag_f16_strided(const _Float16* __restrict__ p, int ldb) {
  v16h r;
#pragma unroll
  for (int e = 0; e < 16; ++e) r[e] = p[(size_t)e * ldb];
  return r;
}
// Row-contiguous (B^T access of a row-major W): p = Wrow + kbase + ((lane>>4)<<4)
__device__ __forceinline__ v16h b_frag_f32_row(const float* __restrict__ p) {
  v16h r;
#pragma unroll
  for (int e = 0; e < 16; ++e) r[e] = (_Float16)p[e];
  return r;
}

__device__ __forceinline__ float sigmf(float x) { return 1.f / (1.f + __expf(-x)); }

// ---------------------------------------------------------------------------
// Kernel 1: XK = A @ Wk + bias   (Wk: 128 x 512 row-major, K=128, N=512)
// A rows addressed as (m/inner)*so + (m%inner)*si  (handles the X[:,0] stride)
// one 16x16 output tile per wave
// ---------------------------------------------------------------------------
__global__ void xk_gemm_kernel(const float* __restrict__ A, int so, int si, int inner,
                               const float* __restrict__ Wk,
                               const float* __restrict__ bias,
                               float* __restrict__ C, int Mtot) {
  const int N = 512, K = 128;
  const int lane = threadIdx.x & 31;
  const int wave = (blockIdx.x * blockDim.x + threadIdx.x) >> 5;
  const int ntiles = N / 16;                    // 32
  const int mt = wave / ntiles;
  const int nt = wave % ntiles;
  if (mt * 16 >= Mtot) return;                  // wave-uniform

  const int nlo = lane & 15;
  const int n = nt * 16 + nlo;
  const int m = mt * 16 + nlo;
  const size_t rowoff = (size_t)(m / inner) * so + (size_t)(m % inner) * si;

  v8f acc = {};
#pragma unroll
  for (int kb = 0; kb < K; kb += 32) {
    v16h af = a_frag_f32(A + rowoff + kb, lane);
    v16h bf = b_frag_f32_strided(Wk + (size_t)(kb + ((lane >> 4) << 4)) * N + n, N);
    acc = wmma_f16(af, bf, acc);
  }
  const float bn = bias[n];
  const int mrow = mt * 16 + ((lane >> 4) << 3);
#pragma unroll
  for (int r = 0; r < 8; ++r) C[(size_t)(mrow + r) * N + n] = acc[r] + bn;
}

// ---------------------------------------------------------------------------
// Kernel 2: persistent skip-LSTM. One workgroup per 16 batch rows (grid=4):
// recurrence is independent across batch rows -> no inter-WG sync, only
// __syncthreads between the WMMA phase and the pointwise phase of each step.
// LDS (288 KB / WGP, under the 320 KB CDNA5 limit):
//   f16 copies of rkernel + kernel2 (staged once -> ds-latency B fragments,
//   zero per-step conversion on the sequential critical path),
//   gate activations (f16), cell state (f32), 16-slot skip ring (f16),
//   gate/skip pre-activations (f32).
// Per step: 40 WMMA tiles (4 gates x 8 ntiles + 8 skip tiles) over 8 waves.
// Encoder writes only the final h (fp32, feeds next GEMM); decoder writes the
// full H sequence directly in f16 (feeds the dense WMMA A-side).
// ---------------------------------------------------------------------------
__global__ void skip_lstm_kernel(const float* __restrict__ XK, int xk_b_stride,
                                 int xk_t_stride,
                                 const float* __restrict__ Rk,    // 128 x 512
                                 const float* __restrict__ K2,    // 128 x 128
                                 const float* __restrict__ bias2, // 128
                                 const float* __restrict__ s0p,
                                 float* __restrict__ HoutLast,    // 64x128 fp32 (or null)
                                 _Float16* __restrict__ HoutSeq,  // 64x16384 f16 (or null)
                                 int T) {
  extern __shared__ char smem[];
  _Float16* sRk   = (_Float16*)smem;            // 128*512 f16
  _Float16* sK2   = sRk + 128 * 512;            // 128*128 f16
  _Float16* sH    = sK2 + 128 * 128;            // 16*512  gates (i,f,c,o) f16
  _Float16* sRing = sH + 16 * 512;              // 16 slots * 16 * 128 f16
  float*    sC    = (float*)(sRing + 16 * 16 * 128); // 16*128 cell f32
  float*    sPre  = sC + 16 * 128;              // 16*512 gate preacts f32
  float*    sPreS = sPre + 16 * 512;            // 16*128 skip preacts f32

  const int tid = threadIdx.x;
  const int lane = tid & 31;
  const int wave = tid >> 5;                    // 0..7
  const int bBase = blockIdx.x * 16;
  const float s0 = s0p[0];

  // stage weights to LDS in f16 (once), zero recurrent state
  for (int i = tid; i < 128 * 512; i += blockDim.x) sRk[i] = (_Float16)Rk[i];
  for (int i = tid; i < 128 * 128; i += blockDim.x) sK2[i] = (_Float16)K2[i];
  for (int i = tid; i < 16 * 512; i += blockDim.x) sH[i] = (_Float16)0.f;
  for (int i = tid; i < 16 * 16 * 128; i += blockDim.x) sRing[i] = (_Float16)0.f;
  for (int i = tid; i < 16 * 128; i += blockDim.x) sC[i] = 0.f;
  __syncthreads();

  const int nlo = lane & 15;
  for (int t = 0; t < T; ++t) {
    const int slot = t & 15;
    // ---- WMMA phase: wave handles 5 of the 40 16x16 tiles --------------
#pragma unroll
    for (int j = 0; j < 5; ++j) {
      const int tile = wave * 5 + j;            // wave-uniform
      v8f acc = {};
      if (tile < 32) {                          // gate tiles: h[:,g] @ Rk[:,g]
        const int g = tile >> 3;
        const int nt = tile & 7;
        const int n = nt * 16 + nlo;
        const _Float16* Arow = sH + nlo * 512 + g * 128;
#pragma unroll
        for (int kb = 0; kb < 128; kb += 32) {
          v16h af = a_frag_f16(Arow + kb, lane);
          v16h bf = b_frag_f16_strided(
              sRk + (size_t)(kb + ((lane >> 4) << 4)) * 512 + g * 128 + n, 512);
          acc = wmma_f16(af, bf, acc);
        }
        const int mrow = (lane >> 4) << 3;
#pragma unroll
        for (int r = 0; r < 8; ++r) sPre[(mrow + r) * 512 + g * 128 + n] = acc[r];
      } else {                                  // skip tiles: prev_h[:, :u] @ K2
        const int nt = tile - 32;
        const int n = nt * 16 + nlo;
        const _Float16* Arow = sRing + slot * 2048 + nlo * 128;
#pragma unroll
        for (int kb = 0; kb < 128; kb += 32) {
          v16h af = a_frag_f16(Arow + kb, lane);
          v16h bf = b_frag_f16_strided(
              sK2 + (size_t)(kb + ((lane >> 4) << 4)) * 128 + n, 128);
          acc = wmma_f16(af, bf, acc);
        }
        const int mrow = (lane >> 4) << 3;
#pragma unroll
        for (int r = 0; r < 8; ++r) sPreS[(mrow + r) * 128 + n] = acc[r];
      }
    }
    __syncthreads();

    // ---- pointwise phase: 2048 cells / 256 threads ---------------------
#pragma unroll
    for (int q = 0; q < 8; ++q) {
      const int idx = tid + q * 256;
      const int m = idx >> 7;
      const int n = idx & 127;
      const int b = bBase + m;
      const float* xk = XK + (size_t)b * xk_b_stride + (size_t)t * xk_t_stride;
      const float xi = xk[n], xf = xk[128 + n], xc = xk[256 + n], xo = xk[384 + n];
      const float iG = sigmf(xi + sPre[m * 512 + n]);
      const float fG = sigmf(xf + sPre[m * 512 + 128 + n]);
      const float cG = fG * sC[m * 128 + n] + iG * tanhf(xc + sPre[m * 512 + 256 + n]);
      const float oG = sigmf(xo + sPre[m * 512 + 384 + n]);
      const float hcell = oG * tanhf(cG);
      const float skip = sigmf(sPreS[m * 128 + n] + bias2[n]);
      const float h = s0 * hcell + skip * (1.f - s0);
      sH[m * 512 + n]       = (_Float16)iG;   // new carry = concat(i,f,c,o)
      sH[m * 512 + 128 + n] = (_Float16)fG;
      sH[m * 512 + 256 + n] = (_Float16)cG;
      sH[m * 512 + 384 + n] = (_Float16)oG;
      sC[m * 128 + n] = cG;
      sRing[slot * 2048 + m * 128 + n] = (_Float16)h;  // append to skip ring
      if (HoutSeq)
        HoutSeq[(size_t)b * 16384 + t * 128 + n] = (_Float16)h;
      else if (t == T - 1)
        HoutLast[b * 128 + n] = h;
    }
    __syncthreads();
  }
}

// ---------------------------------------------------------------------------
// Kernel 3: Y(64x16384) = flat(64x16384) @ W^T + b.   Memory-bound on the
// 1 GB fp32 W stream -> each wave owns 16 output columns and accumulates all
// 64 batch rows so W is read exactly once (no M-tile reuse penalty).
// flat arrives in f16 from the decoder: A fragments load directly with no
// per-iteration conversion, halving the L2-side activation traffic.
// ---------------------------------------------------------------------------
__global__ void dense_kernel(const _Float16* __restrict__ Aflat, // 64 x 16384 f16
                             const float* __restrict__ W,        // 16384 x 16384
                             const float* __restrict__ bias,
                             float* __restrict__ Y) {            // 64 x 16384
  const int K = 16384, N = 16384;
  const int lane = threadIdx.x & 31;
  const int wave = (blockIdx.x * blockDim.x + threadIdx.x) >> 5; // n-tile id
  const int nlo = lane & 15;
  const int n = wave * 16 + nlo;
  const float* wrow = W + (size_t)n * K;

  v8f acc0 = {}, acc1 = {}, acc2 = {}, acc3 = {};
  for (int kb = 0; kb < K; kb += 32) {
    __builtin_prefetch(wrow + kb + 1024, 0, 1);   // global_prefetch_b8 on W stream
    v16h bf = b_frag_f32_row(wrow + kb + ((lane >> 4) << 4));
    v16h a0 = a_frag_f16(Aflat + (size_t)(nlo +  0) * K + kb, lane);
    v16h a1 = a_frag_f16(Aflat + (size_t)(nlo + 16) * K + kb, lane);
    v16h a2 = a_frag_f16(Aflat + (size_t)(nlo + 32) * K + kb, lane);
    v16h a3 = a_frag_f16(Aflat + (size_t)(nlo + 48) * K + kb, lane);
    acc0 = wmma_f16(a0, bf, acc0);
    acc1 = wmma_f16(a1, bf, acc1);
    acc2 = wmma_f16(a2, bf, acc2);
    acc3 = wmma_f16(a3, bf, acc3);
  }
  const float bn = bias[n];
  const int mrow = (lane >> 4) << 3;
#pragma unroll
  for (int r = 0; r < 8; ++r) {
    Y[(size_t)(mrow + r +  0) * N + n] = acc0[r] + bn;
    Y[(size_t)(mrow + r + 16) * N + n] = acc1[r] + bn;
    Y[(size_t)(mrow + r + 32) * N + n] = acc2[r] + bn;
    Y[(size_t)(mrow + r + 48) * N + n] = acc3[r] + bn;
  }
}

// ---------------------------------------------------------------------------
extern "C" void kernel_launch(void* const* d_in, const int* in_sizes, int n_in,
                              void* d_out, int out_size, void* d_ws, size_t ws_size,
                              hipStream_t stream) {
  (void)in_sizes; (void)n_in; (void)out_size; (void)ws_size;
  const float* X   = (const float*)d_in[0];   // (64, 2, 128, 128)
  const float* eK  = (const float*)d_in[1];
  const float* eR  = (const float*)d_in[2];
  const float* eK2 = (const float*)d_in[3];
  const float* eB  = (const float*)d_in[4];
  const float* eB2 = (const float*)d_in[5];
  const float* eS0 = (const float*)d_in[6];
  const float* dK  = (const float*)d_in[7];
  const float* dR  = (const float*)d_in[8];
  const float* dK2 = (const float*)d_in[9];
  const float* dB  = (const float*)d_in[10];
  const float* dB2 = (const float*)d_in[11];
  const float* dS0 = (const float*)d_in[12];
  const float* Wd  = (const float*)d_in[13];
  const float* bd  = (const float*)d_in[14];
  float* out = (float*)d_out;

  float*    ws  = (float*)d_ws;
  float*    XKe = ws;                       // 8192*512 fp32 (16 MB)
  float*    RE  = XKe + 8192 * 512;         // 64*128 fp32
  float*    XKd = RE + 64 * 128;            // 64*512 fp32
  _Float16* FLT = (_Float16*)(XKd + 64 * 512); // 64*16384 f16 (2 MB)

  const size_t smem =
      (size_t)(128 * 512 + 128 * 128 + 16 * 512 + 16 * 16 * 128) * sizeof(_Float16) +
      (size_t)(16 * 128 + 16 * 512 + 16 * 128) * sizeof(float);   // 288 KB
  (void)hipFuncSetAttribute((const void*)skip_lstm_kernel,
                            hipFuncAttributeMaxDynamicSharedMemorySize, (int)smem);

  // 1) encoder input GEMM over all timesteps: XKe[b*128+t] = X[b,0,t]@eK + eB
  xk_gemm_kernel<<<2048, 256, 0, stream>>>(X, 2 * 128 * 128, 128, 128, eK, eB,
                                           XKe, 8192);
  // 2) encoder recurrence -> RE (last h only, fp32)
  skip_lstm_kernel<<<4, 256, smem, stream>>>(XKe, 128 * 512, 512, eR, eK2, eB2,
                                             eS0, RE, ((_Float16*)0), 128);
  // 3) decoder input GEMM (input is RE broadcast over T -> compute once)
  xk_gemm_kernel<<<16, 256, 0, stream>>>(RE, 128, 0, 1, dK, dB, XKd, 64);
  // 4) decoder recurrence -> FLT = full H sequence in f16 (b, t*128+u)
  skip_lstm_kernel<<<4, 256, smem, stream>>>(XKd, 512, 0, dR, dK2, dB2, dS0,
                                             ((float*)0), FLT, 128);
  // 5) dense layer: out = FLT @ Wd^T + bd
  dense_kernel<<<128, 256, 0, stream>>>(FLT, Wd, bd, out);
}
